// mil_outputs_86285892976832
// MI455X (gfx1250) — compile-verified, matrix operand order checked by reading, not run on hardware
//
#include <hip/hip_runtime.h>
#include <hip/hip_bf16.h>
#include <stdint.h>

typedef __bf16 bf16_t;
typedef __attribute__((ext_vector_type(8)))  __bf16 v8bf;
typedef __attribute__((ext_vector_type(16))) __bf16 v16bf;
typedef __attribute__((ext_vector_type(8)))  float  v8f;

#define NEG_SLOPE 0.2f

// ---------------------------------------------------------------------------
// Fragment load: 16 bf16 per lane = two 16-byte global_load_b128 chunks.
// Lane L (0..15): row = L, K offs {0..7} in elems 0..7, {16..23} in 8..15.
// Lane L (16..31): row = L-16, K offs {8..15} / {24..31}.  Caller passes
// p = base + row*ld + kbase + ((lane>>4)*8).
// ---------------------------------------------------------------------------
__device__ __forceinline__ v16bf load_frag16(const bf16_t* p) {
  v8bf lo = *(const v8bf*)(p);
  v8bf hi = *(const v8bf*)(p + 16);
  v16bf f;
#pragma unroll
  for (int i = 0; i < 8; ++i) { f[i] = lo[i]; f[i + 8] = hi[i]; }
  return f;
}

// ---------------------------------------------------------------------------
// Elementwise conversions / packing
// ---------------------------------------------------------------------------
__global__ void f32_to_bf16_kernel(const float* __restrict__ s,
                                   bf16_t* __restrict__ d, long long n) {
  long long i = (long long)blockIdx.x * blockDim.x + threadIdx.x;
  long long st = (long long)gridDim.x * blockDim.x;
  for (; i < n; i += st) d[i] = (bf16_t)s[i];
}

// Pack s0 into rows [0,rows), zeros to blockRows; optionally s1 into
// rows [blockRows, blockRows+rows), zeros to 2*blockRows.
__global__ void pack_pair_bf16_kernel(const float* __restrict__ s0,
                                      const float* __restrict__ s1,
                                      bf16_t* __restrict__ d,
                                      int rows, int K, int blockRows) {
  long long total = (long long)(s1 ? 2 : 1) * blockRows * K;
  long long i = (long long)blockIdx.x * blockDim.x + threadIdx.x;
  long long st = (long long)gridDim.x * blockDim.x;
  for (; i < total; i += st) {
    int r = (int)(i / K), c = (int)(i % K);
    float v = 0.f;
    if (r < rows)                       v = s0[(size_t)r * K + c];
    else if (s1 && r >= blockRows && r < blockRows + rows)
                                        v = s1[(size_t)(r - blockRows) * K + c];
    d[i] = (bf16_t)v;
  }
}

__global__ void zero_u32_kernel(unsigned* __restrict__ p, long long n) {
  long long i = (long long)blockIdx.x * blockDim.x + threadIdx.x;
  long long st = (long long)gridDim.x * blockDim.x;
  for (; i < n; i += st) p[i] = 0u;
}

// ---------------------------------------------------------------------------
// Big GEMM (layer 1): C[M,2*hid] = A[M,K](bf16) x B[2*hid,K](bf16)^T
// 8 waves/block; each wave owns a 16x64 slice (4 WMMA tiles), A-frag reused 4x.
// K loop unrolled 2x with explicit ping-pong fragment sets: no rotating-buffer
// register copies; each set's refill loads stay in flight for a full
// half-iteration before the WMMAs that consume them (partial s_wait_loadcnt).
// Requires K % 64 == 0 (holds: 4096, 1024).
// ---------------------------------------------------------------------------
__global__ __launch_bounds__(256) void gemm1_kernel(
    const bf16_t* __restrict__ A, const bf16_t* __restrict__ B,
    float* __restrict__ hsrc, float* __restrict__ hdst,
    int M, int K, int hid) {
  const int lane  = threadIdx.x & 31;
  const int wave  = threadIdx.x >> 5;
  const int mbase = blockIdx.x * 16;
  const int nbase = blockIdx.y * 512 + wave * 64;
  int arow = mbase + (lane & 15); if (arow >= M) arow = M - 1;
  const int koffl = (lane >> 4) * 8;

  const bf16_t* ap = A + (size_t)arow * K + koffl;
  const bf16_t* bp[4];
#pragma unroll
  for (int t = 0; t < 4; ++t)
    bp[t] = B + (size_t)(nbase + t * 16 + (lane & 15)) * K + koffl;

  v8f acc[4] = {};

  // prologue: two chunk-sets (k=0 and k=32)
  v16bf aA = load_frag16(ap);
  v16bf bA[4];
#pragma unroll
  for (int t = 0; t < 4; ++t) bA[t] = load_frag16(bp[t]);
  v16bf aB = load_frag16(ap + 32);
  v16bf bB[4];
#pragma unroll
  for (int t = 0; t < 4; ++t) bB[t] = load_frag16(bp[t] + 32);

  for (int k = 64; k < K; k += 64) {
    // consume set A (loads completed >= one half-iteration ago)
#pragma unroll
    for (int t = 0; t < 4; ++t)
      acc[t] = __builtin_amdgcn_wmma_f32_16x16x32_bf16(
          false, aA, false, bA[t], (short)0, acc[t], false, false);
    // refill set A in place (chunk k)
    aA = load_frag16(ap + k);
    __builtin_prefetch(ap + k + 128, 0, 1);  // global_prefetch_b8
#pragma unroll
    for (int t = 0; t < 4; ++t) bA[t] = load_frag16(bp[t] + k);

    // consume set B
#pragma unroll
    for (int t = 0; t < 4; ++t)
      acc[t] = __builtin_amdgcn_wmma_f32_16x16x32_bf16(
          false, aB, false, bB[t], (short)0, acc[t], false, false);
    // refill set B in place (chunk k+32)
    aB = load_frag16(ap + k + 32);
#pragma unroll
    for (int t = 0; t < 4; ++t) bB[t] = load_frag16(bp[t] + k + 32);
  }
  // epilogue: consume both sets
#pragma unroll
  for (int t = 0; t < 4; ++t)
    acc[t] = __builtin_amdgcn_wmma_f32_16x16x32_bf16(
        false, aA, false, bA[t], (short)0, acc[t], false, false);
#pragma unroll
  for (int t = 0; t < 4; ++t)
    acc[t] = __builtin_amdgcn_wmma_f32_16x16x32_bf16(
        false, aB, false, bB[t], (short)0, acc[t], false, false);

  const int mrow0 = mbase + (lane >> 4) * 8;
#pragma unroll
  for (int t = 0; t < 4; ++t) {
    int ncol = nbase + t * 16 + (lane & 15);
    float* dst; int c;
    if (ncol < hid) { dst = hsrc; c = ncol; } else { dst = hdst; c = ncol - hid; }
#pragma unroll
    for (int r = 0; r < 8; ++r) {
      int m = mrow0 + r;
      if (m < M) dst[(size_t)m * hid + c] = acc[t][r];
    }
  }
}

// ---------------------------------------------------------------------------
// Generic one-tile-per-wave GEMM with optional bias and column mask.
// grid.x = M tiles, grid.y = N tiles, block = 32 (one wave).
// Same 2x-unrolled ping-pong pipeline; requires K % 64 == 0.
// ---------------------------------------------------------------------------
__global__ __launch_bounds__(32) void gemm_tile_kernel(
    const bf16_t* __restrict__ A, const bf16_t* __restrict__ B,
    const float* __restrict__ bias, float* __restrict__ C,
    int M, int K, int ldc, int ncolLimit) {
  const int lane  = threadIdx.x & 31;
  const int mbase = blockIdx.x * 16;
  const int nbase = blockIdx.y * 16;
  int arow = mbase + (lane & 15); if (arow >= M) arow = M - 1;
  const int brow  = nbase + (lane & 15);
  const int koffl = (lane >> 4) * 8;

  const bf16_t* ap = A + (size_t)arow * K + koffl;
  const bf16_t* bp = B + (size_t)brow * K + koffl;

  v8f acc = {};
  v16bf aA = load_frag16(ap);
  v16bf bA = load_frag16(bp);
  v16bf aB = load_frag16(ap + 32);
  v16bf bB = load_frag16(bp + 32);
  for (int k = 64; k < K; k += 64) {
    acc = __builtin_amdgcn_wmma_f32_16x16x32_bf16(
        false, aA, false, bA, (short)0, acc, false, false);
    aA = load_frag16(ap + k);
    bA = load_frag16(bp + k);
    acc = __builtin_amdgcn_wmma_f32_16x16x32_bf16(
        false, aB, false, bB, (short)0, acc, false, false);
    aB = load_frag16(ap + k + 32);
    bB = load_frag16(bp + k + 32);
  }
  acc = __builtin_amdgcn_wmma_f32_16x16x32_bf16(
      false, aA, false, bA, (short)0, acc, false, false);
  acc = __builtin_amdgcn_wmma_f32_16x16x32_bf16(
      false, aB, false, bB, (short)0, acc, false, false);

  const int mrow0 = mbase + (lane >> 4) * 8;
  const int ncol  = nbase + (lane & 15);
  if (ncol < ncolLimit) {
    float bv = bias ? bias[ncol] : 0.f;
#pragma unroll
    for (int r = 0; r < 8; ++r) {
      int m = mrow0 + r;
      if (m < M) C[(size_t)m * ldc + ncol] = acc[r] + bv;
    }
  }
}

// ---------------------------------------------------------------------------
// Attention coefficients, layer 1: one wave per (node, src/dst) pair.
// ---------------------------------------------------------------------------
__global__ __launch_bounds__(64) void attn1_kernel(
    const float* __restrict__ hsrc, const float* __restrict__ hdst,
    const float* __restrict__ as1,  const float* __restrict__ ad1,
    float* __restrict__ asrc, float* __restrict__ adst, int hid) {
  const int n = blockIdx.x;
  const int lane = threadIdx.x & 31;
  const bool isD = threadIdx.x >= 32;
  const float* h = isD ? hdst : hsrc;
  const float* a = isD ? ad1 : as1;
  float s = 0.f;
  for (int c = lane; c < hid; c += 32) s += h[(size_t)n * hid + c] * a[c];
#pragma unroll
  for (int m = 16; m > 0; m >>= 1) s += __shfl_xor(s, m, 32);
  if (lane == 0) (isD ? adst : asrc)[n] = s;
}

__global__ void attn2_kernel(const float* __restrict__ h2,
                             const float* __restrict__ as2,
                             const float* __restrict__ ad2,
                             float* __restrict__ a2s, float* __restrict__ a2d,
                             int N, int C) {
  int n = blockIdx.x * blockDim.x + threadIdx.x;
  if (n >= N) return;
  float s = 0.f, d = 0.f;
  for (int c = 0; c < C; ++c) {
    s += h2[(size_t)n * 64 + c] * as2[c];
    d += h2[(size_t)n * 64 + 32 + c] * ad2[c];
  }
  a2s[n] = s; a2d[n] = d;
}

// ---------------------------------------------------------------------------
// Edge softmax (monotone-uint encoding for float atomicMax)
// ---------------------------------------------------------------------------
__device__ __forceinline__ unsigned fenc(float f) {
  unsigned u = __float_as_uint(f);
  return (u & 0x80000000u) ? ~u : (u | 0x80000000u);
}
__device__ __forceinline__ float fdec(unsigned e) {
  unsigned u = (e & 0x80000000u) ? (e & 0x7fffffffu) : ~e;
  return __uint_as_float(u);
}

__global__ void edge_logits_kernel(const int* __restrict__ edges, int E, int EE,
                                   const float* __restrict__ asrc,
                                   const float* __restrict__ adst,
                                   float* __restrict__ ebuf,
                                   unsigned* __restrict__ menc) {
  int i = blockIdx.x * blockDim.x + threadIdx.x;
  if (i >= EE) return;
  int s = (i < E) ? edges[i]     : (i - E);
  int d = (i < E) ? edges[E + i] : (i - E);
  float v = asrc[s] + adst[d];
  v = (v >= 0.f) ? v : NEG_SLOPE * v;
  ebuf[i] = v;
  atomicMax(&menc[d], fenc(v));
}

__global__ void edge_exp_kernel(const int* __restrict__ edges, int E, int EE,
                                const float* __restrict__ ebuf,
                                const unsigned* __restrict__ menc,
                                float* __restrict__ exbuf,
                                float* __restrict__ denom) {
  int i = blockIdx.x * blockDim.x + threadIdx.x;
  if (i >= EE) return;
  int d = (i < E) ? edges[E + i] : (i - E);
  float x = __expf(ebuf[i] - fdec(menc[d]));
  exbuf[i] = x;
  atomicAdd(&denom[d], x);
}

// layer-1 aggregation: one 256-thread block per edge, 1024 features
__global__ __launch_bounds__(256) void agg1_kernel(
    const int* __restrict__ edges, int E, int EE,
    const float* __restrict__ exbuf, const float* __restrict__ denom,
    const float* __restrict__ hsrc, float* __restrict__ agg, int hid) {
  int i = blockIdx.x;
  if (i >= EE) return;
  int s = (i < E) ? edges[i]     : (i - E);
  int d = (i < E) ? edges[E + i] : (i - E);
  float coef = exbuf[i] / (denom[d] + 1e-16f);
  const float* hrow = hsrc + (size_t)s * hid;
  float* arow = agg + (size_t)d * hid;
  for (int c = threadIdx.x; c < hid; c += blockDim.x)
    atomicAdd(&arow[c], coef * hrow[c]);
}

// layer-2 aggregation: flat over (edge, feature)
__global__ void agg2_kernel(const int* __restrict__ edges, int E, int EE,
                            const float* __restrict__ exbuf,
                            const float* __restrict__ denom,
                            const float* __restrict__ h2,
                            float* __restrict__ agg, int C) {
  long long idx = (long long)blockIdx.x * blockDim.x + threadIdx.x;
  if (idx >= (long long)EE * C) return;
  int i = (int)(idx / C), c = (int)(idx % C);
  int s = (i < E) ? edges[i]     : (i - E);
  int d = (i < E) ? edges[E + i] : (i - E);
  float coef = exbuf[i] / (denom[d] + 1e-16f);
  atomicAdd(&agg[(size_t)d * C + c], coef * h2[(size_t)s * 64 + c]);
}

// bias + ELU + convert to bf16 for the layer-2 GEMM
__global__ void elu_bf16_kernel(const float* __restrict__ agg,
                                const float* __restrict__ b1,
                                bf16_t* __restrict__ hb, long long n, int hid) {
  long long i = (long long)blockIdx.x * blockDim.x + threadIdx.x;
  long long st = (long long)gridDim.x * blockDim.x;
  for (; i < n; i += st) {
    float v = agg[i] + b1[i % hid];
    v = (v > 0.f) ? v : (__expf(v) - 1.f);
    hb[i] = (bf16_t)v;
  }
}

// per-class (axis-0) softmax stats: one block per class
__global__ __launch_bounds__(256) void colsoftmax_stats_kernel(
    const float* __restrict__ sc, int N, int C,
    float* __restrict__ cmax, float* __restrict__ csum) {
  __shared__ float red[256];
  int c = blockIdx.x;
  float m = -3.4e38f;
  for (int n = threadIdx.x; n < N; n += blockDim.x)
    m = fmaxf(m, sc[(size_t)n * C + c]);
  red[threadIdx.x] = m; __syncthreads();
  for (int s = 128; s > 0; s >>= 1) {
    if (threadIdx.x < s) red[threadIdx.x] = fmaxf(red[threadIdx.x], red[threadIdx.x + s]);
    __syncthreads();
  }
  m = red[0]; __syncthreads();
  float sum = 0.f;
  for (int n = threadIdx.x; n < N; n += blockDim.x)
    sum += __expf(sc[(size_t)n * C + c] - m);
  red[threadIdx.x] = sum; __syncthreads();
  for (int s = 128; s > 0; s >>= 1) {
    if (threadIdx.x < s) red[threadIdx.x] += red[threadIdx.x + s];
    __syncthreads();
  }
  if (threadIdx.x == 0) { cmax[c] = m; csum[c] = red[0]; }
}

__global__ void final_kernel(const float* __restrict__ sc0,
                             const float* __restrict__ agg2,
                             const float* __restrict__ b2,
                             const float* __restrict__ cmax,
                             const float* __restrict__ csum,
                             float* __restrict__ out, int N, int C) {
  int i = blockIdx.x * blockDim.x + threadIdx.x;
  if (i >= N * C) return;
  int c = i % C;
  float soft = __expf(sc0[i] - cmax[c]) / csum[c];
  out[i] = soft * (agg2[i] + b2[c]);
}

// Emit new_edge (edges + self loops) into output tail.
__global__ void edges_out_kernel(const int* __restrict__ edges, int E, int EE,
                                 void* __restrict__ tail, int mode) {
  int i = blockIdx.x * blockDim.x + threadIdx.x;
  if (i >= EE || mode < 0) return;
  int s = (i < E) ? edges[i]     : (i - E);
  int d = (i < E) ? edges[E + i] : (i - E);
  if (mode == 1) { long long* o = (long long*)tail; o[i] = s; o[EE + i] = d; }
  else           { int*       o = (int*)tail;       o[i] = s; o[EE + i] = d; }
}

// ---------------------------------------------------------------------------
extern "C" void kernel_launch(void* const* d_in, const int* in_sizes, int n_in,
                              void* d_out, int out_size, void* d_ws, size_t ws_size,
                              hipStream_t stream) {
  const float* x    = (const float*)d_in[0];
  const int*   edges= (const int*)  d_in[1];
  const float* W0   = (const float*)d_in[2];
  const float* b0   = (const float*)d_in[3];
  const float* Ws1  = (const float*)d_in[4];
  const float* Wd1  = (const float*)d_in[5];
  const float* as1  = (const float*)d_in[6];
  const float* ad1  = (const float*)d_in[7];
  const float* b1   = (const float*)d_in[8];
  const float* Ws2  = (const float*)d_in[9];
  const float* Wd2  = (const float*)d_in[10];
  const float* as2  = (const float*)d_in[11];
  const float* ad2  = (const float*)d_in[12];
  const float* b2   = (const float*)d_in[13];

  const int C   = in_sizes[3];            // 20
  const int Kin = in_sizes[2] / C;        // 4096
  const int N   = in_sizes[0] / Kin;      // 10000
  const int E   = in_sizes[1] / 2;        // 160000
  const int hid = in_sizes[8];            // 1024
  const int EE  = E + N;                  // 170000

  // ---- workspace carve (aligned 512B) ----
  char* wp = (char*)d_ws;
  auto alloc = [&](size_t bytes) -> void* {
    void* p = (void*)wp; wp += (bytes + 511) & ~(size_t)511; return p;
  };
  bf16_t*   xb    = (bf16_t*)  alloc((size_t)N * Kin * 2);
  bf16_t*   w1b   = (bf16_t*)  alloc((size_t)2 * hid * Kin * 2);
  bf16_t*   w2b   = (bf16_t*)  alloc((size_t)64 * hid * 2);
  bf16_t*   w0b   = (bf16_t*)  alloc((size_t)32 * Kin * 2);
  float*    hsrc  = (float*)   alloc((size_t)N * hid * 4);
  float*    hdst  = (float*)   alloc((size_t)N * hid * 4);  // reused as agg1
  bf16_t*   hb    = (bf16_t*)  alloc((size_t)N * hid * 2);
  float*    asrc  = (float*)   alloc((size_t)N * 4);
  float*    adst  = (float*)   alloc((size_t)N * 4);
  float*    ebuf  = (float*)   alloc((size_t)EE * 4);
  float*    exbuf = (float*)   alloc((size_t)EE * 4);
  unsigned* menc  = (unsigned*)alloc((size_t)N * 4);
  float*    denom = (float*)   alloc((size_t)N * 4);
  float*    h2cat = (float*)   alloc((size_t)N * 64 * 4);
  float*    a2s   = (float*)   alloc((size_t)N * 4);
  float*    a2d   = (float*)   alloc((size_t)N * 4);
  float*    agg2  = (float*)   alloc((size_t)N * C * 4);
  float*    score0= (float*)   alloc((size_t)N * C * 4);
  float*    cmax  = (float*)   alloc(64 * 4);
  float*    csum  = (float*)   alloc(64 * 4);
  float*    agg1  = hdst;  // hdst no longer needed after attn1

  const int Mt = (N + 15) / 16;
  dim3 b256(256);

  // ---- precision conversion / weight packing ----
  f32_to_bf16_kernel<<<2048, b256, 0, stream>>>(x,   xb,  (long long)N * Kin);
  f32_to_bf16_kernel<<<1024, b256, 0, stream>>>(Ws1, w1b, (long long)hid * Kin);
  f32_to_bf16_kernel<<<1024, b256, 0, stream>>>(Wd1, w1b + (size_t)hid * Kin,
                                                (long long)hid * Kin);
  pack_pair_bf16_kernel<<<256, b256, 0, stream>>>(Ws2, Wd2, w2b, C, hid, 32);
  pack_pair_bf16_kernel<<<256, b256, 0, stream>>>(W0, nullptr, w0b, C, Kin, 32);

  // ---- mil_score0 = x @ W0^T + b0 (WMMA) ----
  gemm_tile_kernel<<<dim3(Mt, 2), 32, 0, stream>>>(xb, w0b, b0, score0,
                                                   N, Kin, C, C);

  // ---- layer-1 projections (dominant GEMM, WMMA bf16) ----
  gemm1_kernel<<<dim3(Mt, (2 * hid) / 512), 256, 0, stream>>>(
      xb, w1b, hsrc, hdst, N, Kin, hid);

  // ---- layer-1 attention + edge softmax + aggregation ----
  attn1_kernel<<<N, 64, 0, stream>>>(hsrc, hdst, as1, ad1, asrc, adst, hid);
  zero_u32_kernel<<<64,  b256, 0, stream>>>(menc, N);
  zero_u32_kernel<<<64,  b256, 0, stream>>>((unsigned*)denom, N);
  zero_u32_kernel<<<2048, b256, 0, stream>>>((unsigned*)agg1, (long long)N * hid);
  edge_logits_kernel<<<(EE + 255) / 256, b256, 0, stream>>>(edges, E, EE,
                                                            asrc, adst, ebuf, menc);
  edge_exp_kernel<<<(EE + 255) / 256, b256, 0, stream>>>(edges, E, EE,
                                                         ebuf, menc, exbuf, denom);
  agg1_kernel<<<EE, b256, 0, stream>>>(edges, E, EE, exbuf, denom, hsrc, agg1, hid);
  elu_bf16_kernel<<<2048, b256, 0, stream>>>(agg1, b1, hb, (long long)N * hid, hid);

  // ---- layer-2 projections (WMMA, padded C=20 -> tiles of 16) ----
  gemm_tile_kernel<<<dim3(Mt, 4), 32, 0, stream>>>(hb, w2b, nullptr, h2cat,
                                                   N, hid, 64, 64);
  attn2_kernel<<<(N + 255) / 256, b256, 0, stream>>>(h2cat, as2, ad2, a2s, a2d, N, C);

  // ---- layer-2 edge softmax + aggregation ----
  zero_u32_kernel<<<64, b256, 0, stream>>>(menc, N);
  zero_u32_kernel<<<64, b256, 0, stream>>>((unsigned*)denom, N);
  zero_u32_kernel<<<64, b256, 0, stream>>>((unsigned*)agg2, (long long)N * C);
  edge_logits_kernel<<<(EE + 255) / 256, b256, 0, stream>>>(edges, E, EE,
                                                            a2s, a2d, ebuf, menc);
  edge_exp_kernel<<<(EE + 255) / 256, b256, 0, stream>>>(edges, E, EE,
                                                         ebuf, menc, exbuf, denom);
  {
    long long tot = (long long)EE * C;
    agg2_kernel<<<(unsigned)((tot + 255) / 256), b256, 0, stream>>>(
        edges, E, EE, exbuf, denom, h2cat, agg2, C);
  }

  // ---- fused final: softmax(score0, axis=0) * (agg2 + b2) ----
  colsoftmax_stats_kernel<<<C, b256, 0, stream>>>(score0, N, C, cmax, csum);
  final_kernel<<<(N * C + 255) / 256, b256, 0, stream>>>(
      score0, agg2, b2, cmax, csum, (float*)d_out, N, C);

  // ---- emit new_edge into output tail (int64 if room, else int32) ----
  {
    long long base = (long long)N * C;
    int mode = -1;
    if ((long long)out_size >= base + 4LL * EE)      mode = 1;  // int64
    else if ((long long)out_size >= base + 2LL * EE) mode = 0;  // int32
    void* tail = (void*)((float*)d_out + base);
    edges_out_kernel<<<(EE + 255) / 256, b256, 0, stream>>>(edges, E, EE, tail, mode);
  }
}
